// CrossDConv_18451179503778
// MI455X (gfx1250) — matrix operand order repaired and checked by reading, not compiled, not run on hardware
//
#include <hip/hip_runtime.h>
#include <math.h>

typedef __attribute__((ext_vector_type(16))) _Float16 v16h;
typedef __attribute__((ext_vector_type(8)))  _Float16 v8h;
typedef __attribute__((ext_vector_type(2)))  _Float16 v2h;
typedef __attribute__((ext_vector_type(8)))  float    v8f;
typedef __attribute__((ext_vector_type(4)))  float    v4f;

#define MTOT 16384   // B*H*W = 4*64*64
#define HW   4096    // 64*64
#define CIN  64
#define NPTS 52

// ------------------------------------------------------------------
// Pack an (O, 64) f32 weight matrix into (Opad, 64) f16, zero-padded.
// ------------------------------------------------------------------
__global__ __launch_bounds__(256) void pack_w16(
    const float* __restrict__ w, _Float16* __restrict__ wh, int O, int Opad)
{
    int i = blockIdx.x * blockDim.x + threadIdx.x;
    if (i >= Opad * CIN) return;
    int o = i >> 6;
    wh[i] = (o < O) ? (_Float16)w[i] : (_Float16)0.f;
}

// Zero-pad a bias vector from O to Opad entries.
__global__ __launch_bounds__(128) void pack_b32(
    const float* __restrict__ b, float* __restrict__ bp, int O, int Opad)
{
    int i = blockIdx.x * blockDim.x + threadIdx.x;
    if (i < Opad) bp[i] = (i < O) ? b[i] : 0.f;
}

// ------------------------------------------------------------------
// Fused depthwise 3x3 for the three branches.
// Consecutive lanes cover consecutive channels of one position, so the
// packed (M,64) f16 stores are fully coalesced.
// ------------------------------------------------------------------
__global__ __launch_bounds__(256) void dw3_fused(
    const float* __restrict__ x,
    const float* __restrict__ wf, const float* __restrict__ bf,
    const float* __restrict__ wc, const float* __restrict__ bc,
    const float* __restrict__ wm, const float* __restrict__ bm,
    _Float16* __restrict__ tf, _Float16* __restrict__ tc, _Float16* __restrict__ tm)
{
    int idx = blockIdx.x * blockDim.x + threadIdx.x;  // m*64 + c
    int c = idx & 63;
    int m = idx >> 6;
    int b = m >> 12;
    int hw = m & (HW - 1);
    int h = hw >> 6;
    int w = hw & 63;
    const float* xp = x + ((b * CIN + c) << 12);
    float sf = 0.f, sc = 0.f, sm = 0.f;
#pragma unroll
    for (int dy = -1; dy <= 1; ++dy) {
        int yy = h + dy;
        bool vy = (yy >= 0) && (yy < 64);
#pragma unroll
        for (int dx = -1; dx <= 1; ++dx) {
            int xx = w + dx;
            bool v = vy && (xx >= 0) && (xx < 64);
            float val = v ? xp[(yy << 6) + xx] : 0.f;
            int j = (dy + 1) * 3 + (dx + 1);
            sf = fmaf(val, wf[c * 9 + j], sf);
            sc = fmaf(val, wc[c * 9 + j], sc);
            sm = fmaf(val, wm[c * 9 + j], sm);
        }
    }
    tf[idx] = (_Float16)(sf + bf[c]);
    tc[idx] = (_Float16)(sc + bc[c]);
    tm[idx] = (_Float16)(sm + bm[c]);
}

// ------------------------------------------------------------------
// K=64 GEMM on v_wmma_f32_16x16x32_f16 with pre-packed f16 operands.
// Each wave: one 16-channel tile x TWO 16-position tiles sharing the A
// (weight) fragment -> per K-chunk: 2 A-loads + 4 B-loads + 2 WMMAs.
// All operands are padded so the epilogue is branch-free.
// MODE 0: planar f32 rows [row_off, row_off+Opad) of outf (padded buffer)
// MODE 1: packed f16 (M,64) to outh (one b128 store per lane), RELU opt.
// MODE 2: f32 residual add (bchw layout) into outf
// ------------------------------------------------------------------
template<int MODE, int RELU>
__global__ __launch_bounds__(256) void gemm_wmma_k64(
    const _Float16* __restrict__ acth,  // (MTOT, 64) f16
    const _Float16* __restrict__ wgth,  // (Opad, 64) f16 zero-padded
    const float* __restrict__ biasp,    // (Opad) zero-padded
    float* __restrict__ outf,
    _Float16* __restrict__ outh,
    const float* __restrict__ resid,    // MODE 2 only (bchw)
    int row_off)
{
    int lane = threadIdx.x & 31;
    int wave = threadIdx.x >> 5;
    int half = lane >> 4;
    int lr   = lane & 15;
    int m0 = (blockIdx.x * 8 + wave) << 5;   // two adjacent 16-position tiles
    int o0 = blockIdx.y << 4;                // 16-channel tile

    const _Float16* ap  = wgth + ((o0 + lr) << 6) + (half << 4);
    const _Float16* bp0 = acth + ((m0 + lr) << 6) + (half << 4);
    const _Float16* bp1 = bp0 + (16 << 6);

    v8f acc0 = {}, acc1 = {};
#pragma unroll
    for (int kk = 0; kk < CIN; kk += 32) {
        v8h alo = *(const v8h*)(ap + kk);         // K = kk + half*16 + 0..7
        v8h ahi = *(const v8h*)(ap + kk + 8);     // K = kk + half*16 + 8..15
        v8h b0l = *(const v8h*)(bp0 + kk);        // K = kk + half*16 + 0..7
        v8h b0h = *(const v8h*)(bp0 + kk + 8);    // K = kk + half*16 + 8..15
        v8h b1l = *(const v8h*)(bp1 + kk);
        v8h b1h = *(const v8h*)(bp1 + kk + 8);
        v16h afrag = __builtin_shufflevector(alo, ahi, 0,1,2,3,4,5,6,7,8,9,10,11,12,13,14,15);
        v16h bf0   = __builtin_shufflevector(b0l, b0h, 0,1,2,3,4,5,6,7,8,9,10,11,12,13,14,15);
        v16h bf1   = __builtin_shufflevector(b1l, b1h, 0,1,2,3,4,5,6,7,8,9,10,11,12,13,14,15);
        acc0 = __builtin_amdgcn_wmma_f32_16x16x32_f16(
            false, afrag, false, bf0, (short)0, acc0, false, false);
        acc1 = __builtin_amdgcn_wmma_f32_16x16x32_f16(
            false, afrag, false, bf1, (short)0, acc1, false, false);
    }

    // Each lane covers 8 consecutive output channels: vector bias load.
    int ob = o0 + (half << 3);                  // D rows: lanes16-31 hold M=8..15
    v4f blo = *(const v4f*)(biasp + ob);
    v4f bhi = *(const v4f*)(biasp + ob + 4);
    float bv[8] = { blo[0], blo[1], blo[2], blo[3], bhi[0], bhi[1], bhi[2], bhi[3] };

#pragma unroll
    for (int t = 0; t < 2; ++t) {
        const v8f& acc = t ? acc1 : acc0;
        int mcol = m0 + (t << 4) + lr;
        if (MODE == 1) {
            v8h pk;
#pragma unroll
            for (int r = 0; r < 8; ++r) {
                float v = acc[r] + bv[r];
                if (RELU) v = fmaxf(v, 0.f);
                pk[r] = (_Float16)v;
            }
            *(v8h*)(outh + (mcol << 6) + ob) = pk;
        } else if (MODE == 2) {
            int b = mcol >> 12;
            int hw = mcol & (HW - 1);
#pragma unroll
            for (int r = 0; r < 8; ++r) {
                int oi = ((b * CIN + ob + r) << 12) + hw;
                outf[oi] = acc[r] + bv[r] + resid[oi];
            }
        } else {
#pragma unroll
            for (int r = 0; r < 8; ++r)
                outf[(row_off + ob + r) * MTOT + mcol] = acc[r] + bv[r];
        }
    }
}

// ------------------------------------------------------------------
// Sparsity-gated softmax over the 52 point scores.
// ------------------------------------------------------------------
__global__ __launch_bounds__(256) void mask_softmax(
    const float* __restrict__ mraw,  // (64-row padded, rows 0..52 valid)
    float* __restrict__ modb)        // (52, MTOT)
{
    int m = blockIdx.x * blockDim.x + threadIdx.x;
    float sp = mraw[NPTS * MTOT + m];
    float lg[NPTS];
    float mx = -1e30f;
#pragma unroll
    for (int n = 0; n < NPTS; ++n) {
        float s = mraw[n * MTOT + m];
        float sm = 1.f / (1.f + __expf(-(s - sp) * 10.f));   // TAU = 0.1
        float l = s + __logf(sm + 1e-10f);
        lg[n] = l;
        mx = fmaxf(mx, l);
    }
    float sum = 0.f;
#pragma unroll
    for (int n = 0; n < NPTS; ++n) {
        float e = __expf(lg[n] - mx);
        lg[n] = e;
        sum += e;
    }
    float inv = 1.f / sum;
#pragma unroll
    for (int n = 0; n < NPTS; ++n)
        modb[n * MTOT + m] = lg[n] * inv;
}

// ------------------------------------------------------------------
// Bilinear sample (zero padding) + aggregation over the 52 points.
// One wave per position; lane handles channels (2*lane, 2*lane+1) so the
// packed f16 output row is one coalesced 4B store per lane.
// ------------------------------------------------------------------
__global__ __launch_bounds__(256) void sample_agg(
    const float* __restrict__ x, const float* __restrict__ p_n,
    const float* __restrict__ off, const float* __restrict__ modb,
    _Float16* __restrict__ aggh)
{
    int lane = threadIdx.x & 31;
    int wave = threadIdx.x >> 5;
    int m = blockIdx.x * 8 + wave;
    int b = m >> 12;
    int hw = m & (HW - 1);
    int h0 = hw >> 6, w0 = hw & 63;
    const float* xb = x + ((b * CIN) << 12);
    int c0 = lane << 1, c1 = c0 + 1;
    float a0 = 0.f, a1 = 0.f;
    for (int n = 0; n < NPTS; ++n) {
        if (n + 1 < NPTS) {
            __builtin_prefetch(&off[(n + 1) * MTOT + m], 0, 1);
            __builtin_prefetch(&modb[(n + 1) * MTOT + m], 0, 1);
        }
        float wn = modb[n * MTOT + m];
        float cx = (float)w0 + p_n[n]        + off[n * MTOT + m];
        float cy = (float)h0 + p_n[NPTS + n] + off[(NPTS + n) * MTOT + m];
        float fx = floorf(cx), fy = floorf(cy);
        float wx = cx - fx, wy = cy - fy;
        int x0i = (int)fx, y0i = (int)fy;
#pragma unroll
        for (int t = 0; t < 4; ++t) {
            int xx = x0i + (t & 1);
            int yy = y0i + (t >> 1);
            float tw = ((t & 1) ? wx : 1.f - wx) * ((t >> 1) ? wy : 1.f - wy);
            if (xx >= 0 && xx < 64 && yy >= 0 && yy < 64) {
                int base = (yy << 6) + xx;
                float wt = wn * tw;
                a0 = fmaf(wt, xb[(c0 << 12) + base], a0);
                a1 = fmaf(wt, xb[(c1 << 12) + base], a1);
            }
        }
    }
    v2h pk;
    pk[0] = (_Float16)a0;
    pk[1] = (_Float16)a1;
    *(v2h*)(aggh + (m << 6) + c0) = pk;
}

extern "C" void kernel_launch(void* const* d_in, const int* in_sizes, int n_in,
                              void* d_out, int out_size, void* d_ws, size_t ws_size,
                              hipStream_t stream)
{
    (void)in_sizes; (void)n_in; (void)out_size; (void)ws_size;
    const float* x     = (const float*)d_in[0];
    const float* p_n   = (const float*)d_in[1];
    const float* dwf_w = (const float*)d_in[2];
    const float* dwf_b = (const float*)d_in[3];
    const float* pwf_w = (const float*)d_in[4];
    const float* pwf_b = (const float*)d_in[5];
    const float* dwc_w = (const float*)d_in[6];
    const float* dwc_b = (const float*)d_in[7];
    const float* pwc_w = (const float*)d_in[8];
    const float* pwc_b = (const float*)d_in[9];
    const float* dwm_w = (const float*)d_in[10];
    const float* dwm_b = (const float*)d_in[11];
    const float* pwm_w = (const float*)d_in[12];
    const float* pwm_b = (const float*)d_in[13];
    const float* pc_w  = (const float*)d_in[14];
    const float* pc_b  = (const float*)d_in[15];
    const float* w1    = (const float*)d_in[16];
    const float* b1    = (const float*)d_in[17];
    const float* w2    = (const float*)d_in[18];
    const float* b2    = (const float*)d_in[19];
    float* outp = (float*)d_out;

    // ---------------- workspace layout (bytes) ----------------
    char* wsb = (char*)d_ws;
    size_t off_b = 0;
    auto carve = [&](size_t bytes) {
        char* p = wsb + off_b;
        off_b += (bytes + 255) & ~(size_t)255;
        return p;
    };
    _Float16* tfh  = (_Float16*)carve(CIN * MTOT * 2);        // packed dw outputs
    _Float16* tch  = (_Float16*)carve(CIN * MTOT * 2);
    _Float16* tmh  = (_Float16*)carve(CIN * MTOT * 2);
    float*    offb = (float*)   carve((32 + 80) * MTOT * 4);  // padded (112,M); rows 0..103 valid
    float*    mraw = (float*)   carve(64 * MTOT * 4);         // padded (64,M);  rows 0..52 valid
    float*    modb = (float*)   carve(NPTS * MTOT * 4);
    _Float16* aggh = (_Float16*)carve(CIN * MTOT * 2);
    _Float16* outh = (_Float16*)carve(CIN * MTOT * 2);
    _Float16* h1h  = (_Float16*)carve(CIN * MTOT * 2);
    _Float16* pwfh = (_Float16*)carve(32 * CIN * 2);
    _Float16* pwch = (_Float16*)carve(80 * CIN * 2);
    _Float16* pwmh = (_Float16*)carve(64 * CIN * 2);
    _Float16* pch  = (_Float16*)carve(64 * CIN * 2);
    _Float16* w1h  = (_Float16*)carve(64 * CIN * 2);
    _Float16* w2h  = (_Float16*)carve(64 * CIN * 2);
    float*    pwcbp = (float*)  carve(80 * 4);                // padded biases
    float*    pwmbp = (float*)  carve(64 * 4);

    dim3 blk(256);

    // weight / bias packing (f32 -> zero-padded f16 / f32)
    pack_w16<<<dim3((32 * CIN + 255) / 256), blk, 0, stream>>>(pwf_w, pwfh, 32, 32);
    pack_w16<<<dim3((80 * CIN + 255) / 256), blk, 0, stream>>>(pwc_w, pwch, 72, 80);
    pack_w16<<<dim3((64 * CIN + 255) / 256), blk, 0, stream>>>(pwm_w, pwmh, 53, 64);
    pack_w16<<<dim3((64 * CIN + 255) / 256), blk, 0, stream>>>(pc_w,  pch,  64, 64);
    pack_w16<<<dim3((64 * CIN + 255) / 256), blk, 0, stream>>>(w1,    w1h,  64, 64);
    pack_w16<<<dim3((64 * CIN + 255) / 256), blk, 0, stream>>>(w2,    w2h,  64, 64);
    pack_b32<<<dim3(1), dim3(128), 0, stream>>>(pwc_b, pwcbp, 72, 80);
    pack_b32<<<dim3(1), dim3(128), 0, stream>>>(pwm_b, pwmbp, 53, 64);

    // depthwise (all three branches fused), packed f16 output
    dw3_fused<<<dim3((CIN * MTOT) / 256), blk, 0, stream>>>(
        x, dwf_w, dwf_b, dwc_w, dwc_b, dwm_w, dwm_b, tfh, tch, tmh);

    // pointwise convs as WMMA GEMMs -> padded planar f32 (offsets / mask logits)
    // grid.x = MTOT / (32 positions * 8 waves) = 64
    gemm_wmma_k64<0,0><<<dim3(MTOT / 256, 2), blk, 0, stream>>>(tfh, pwfh, pwf_b, offb, nullptr, nullptr, 0);
    gemm_wmma_k64<0,0><<<dim3(MTOT / 256, 5), blk, 0, stream>>>(tch, pwch, pwcbp, offb, nullptr, nullptr, 32);
    gemm_wmma_k64<0,0><<<dim3(MTOT / 256, 4), blk, 0, stream>>>(tmh, pwmh, pwmbp, mraw, nullptr, nullptr, 0);

    // sparsity-gated softmax mask
    mask_softmax<<<dim3(MTOT / 256), blk, 0, stream>>>(mraw, modb);

    // deformable bilinear sampling + aggregation -> packed f16
    sample_agg<<<dim3(MTOT / 8), blk, 0, stream>>>(x, p_n, offb, modb, aggh);

    // pc conv -> packed f16; mlp1 (relu) -> packed f16; mlp2 + residual -> f32 bchw
    gemm_wmma_k64<1,0><<<dim3(MTOT / 256, 4), blk, 0, stream>>>(aggh, pch, pc_b, nullptr, outh, nullptr, 0);
    gemm_wmma_k64<1,1><<<dim3(MTOT / 256, 4), blk, 0, stream>>>(outh, w1h, b1,   nullptr, h1h,  nullptr, 0);
    gemm_wmma_k64<2,0><<<dim3(MTOT / 256, 4), blk, 0, stream>>>(h1h,  w2h, b2,   outp,    nullptr, x,    0);
}
